// CONV_DQRN_30734785970916
// MI455X (gfx1250) — compile-verified, matrix operand order checked by reading, not run on hardware
//
#include <hip/hip_runtime.h>
#include <math.h>

typedef __attribute__((ext_vector_type(16))) __bf16 v16bf;
typedef __attribute__((ext_vector_type(8)))  float  v8f;
typedef __attribute__((ext_vector_type(4)))  unsigned int u32x4;
typedef __attribute__((ext_vector_type(2)))  int i32x2;
typedef __attribute__((ext_vector_type(4)))  int i32x4;

// ---- async global->LDS capability probe (gfx1250 path) ----
#if defined(__has_builtin)
#  if __has_builtin(__builtin_amdgcn_global_load_async_to_lds_b64) && \
      __has_builtin(__builtin_amdgcn_global_load_async_to_lds_b128) && \
      __has_builtin(__builtin_amdgcn_s_wait_asynccnt)
#    define USE_ASYNC_LDS 1
#  endif
#endif
#ifndef USE_ASYNC_LDS
#  define USE_ASYNC_LDS 0
#endif

// builtin signatures: (int2/4 addrspace(1)* gsrc, int2/4 addrspace(3)* ldsdst, imm offset, imm cpol)
typedef __attribute__((address_space(1))) i32x2 g_i32x2;
typedef __attribute__((address_space(3))) i32x2 l_i32x2;
typedef __attribute__((address_space(1))) i32x4 g_i32x4;
typedef __attribute__((address_space(3))) i32x4 l_i32x4;

// ---------- helpers ----------
__device__ __forceinline__ unsigned short f32_to_bf16(float f) {
  unsigned int u = __float_as_uint(f);
  u += 0x7fffu + ((u >> 16) & 1u);           // round-to-nearest-even
  return (unsigned short)(u >> 16);
}
__device__ __forceinline__ float bf16_to_f32(unsigned short h) {
  return __uint_as_float(((unsigned int)h) << 16);
}
__device__ __forceinline__ float sigmoidf_(float x) { return 1.0f / (1.0f + __expf(-x)); }

// Load a 16-element bf16 fragment row (A: row m / B: column n) for one lane.
// Per ISA 7.12.2: elems 0..7 -> K = half+0..7, elems 8..15 -> K = 16+half+0..7,
// caller pre-offsets p by (kb + half).
__device__ __forceinline__ v16bf load_frag16(const __bf16* p) {
  v16bf f;
#pragma unroll
  for (int i = 0; i < 8; i++) { f[i] = p[i]; f[i + 8] = p[16 + i]; }
  return f;
}

__device__ __forceinline__ void wait_async_le2() {
#if USE_ASYNC_LDS
  __builtin_amdgcn_s_wait_asynccnt(2);
#endif
}
__device__ __forceinline__ void wait_async_le0() {
#if USE_ASYNC_LDS
  __builtin_amdgcn_s_wait_asynccnt(0);
#endif
}

// ---------- f32 -> bf16 convert ----------
__global__ void f2bf_kernel(const float* __restrict__ src, unsigned short* __restrict__ dst, int n) {
  int i = blockIdx.x * blockDim.x + threadIdx.x;
  if (i < n) dst[i] = f32_to_bf16(src[i]);
}

// ---------- conv1 (direct, fp32) + relu + 2x2 maxpool -> bf16 ----------
// images (4096,1,28,28) -> x1 (4096,32,12,12) bf16
__global__ void conv1_pool_kernel(const float* __restrict__ img, const float* __restrict__ w,
                                  const float* __restrict__ b, unsigned short* __restrict__ x1) {
  int idx = blockIdx.x * blockDim.x + threadIdx.x;
  if (idx >= 4096 * 32 * 12 * 12) return;
  int pw = idx % 12, ph = (idx / 12) % 12, c = (idx / 144) % 32, n = idx / 4608;
  const float* ip = img + n * 784;
  const float* wp = w + c * 25;
  float bias = b[c];
  float mx = -1e30f;
#pragma unroll
  for (int dh = 0; dh < 2; dh++) {
#pragma unroll
    for (int dw = 0; dw < 2; dw++) {
      int oh = 2 * ph + dh, ow = 2 * pw + dw;
      float s = bias;
#pragma unroll
      for (int kh = 0; kh < 5; kh++)
#pragma unroll
        for (int kw = 0; kw < 5; kw++)
          s = fmaf(ip[(oh + kh) * 28 + (ow + kw)], wp[kh * 5 + kw], s);
      s = fmaxf(s, 0.0f);
      mx = fmaxf(mx, s);
    }
  }
  x1[((n * 32 + c) * 12 + ph) * 12 + pw] = f32_to_bf16(mx);
}

// ---------- conv2 as implicit-im2col WMMA GEMM ----------
// per image: C(64 pos x 64 ch) = A(64x800 im2col of x1) * W2^T(800x64), +bias, relu
// grid = (4 ntiles, 4 mtiles, 4096 imgs), block = 32 (one wave per 16x16 tile)
__global__ void conv2_wmma_kernel(const unsigned short* __restrict__ x1,
                                  const unsigned short* __restrict__ w2,
                                  const float* __restrict__ b2,
                                  unsigned short* __restrict__ y2) {
  const __bf16* xb = reinterpret_cast<const __bf16*>(x1);
  const __bf16* wb = reinterpret_cast<const __bf16*>(w2);
  int lane = threadIdx.x;
  int r    = lane & 15;
  int half = (lane < 16) ? 0 : 8;
  int img  = blockIdx.z;
  int m0   = blockIdx.y * 16;   // output position tile
  int n0   = blockIdx.x * 16;   // output channel tile
  int m = m0 + r, oh = m >> 3, ow = m & 7;
  const __bf16* xbase = xb + img * (32 * 144);
  v8f acc = {};
  for (int kb = 0; kb < 800; kb += 32) {
    v16bf a;
#pragma unroll
    for (int i = 0; i < 8; i++) {
      int k0 = kb + half + i;
      int c0 = k0 / 25, rem0 = k0 % 25, kh0 = rem0 / 5, kw0 = rem0 - kh0 * 5;
      a[i] = xbase[(c0 * 12 + oh + kh0) * 12 + (ow + kw0)];
      int k1 = kb + 16 + half + i;
      int c1 = k1 / 25, rem1 = k1 % 25, kh1 = rem1 / 5, kw1 = rem1 - kh1 * 5;
      a[i + 8] = xbase[(c1 * 12 + oh + kh1) * 12 + (ow + kw1)];
    }
    v16bf bfr = load_frag16(wb + (n0 + r) * 800 + kb + half);
    acc = __builtin_amdgcn_wmma_f32_16x16x32_bf16(false, a, false, bfr, (short)0, acc, false, false);
  }
  float bv = b2[n0 + r];
#pragma unroll
  for (int v = 0; v < 8; v++) {
    int mm = m0 + half + v;                    // C layout: lane=N, vgpr=M (+8 hi lanes)
    float val = fmaxf(acc[v] + bv, 0.0f);
    y2[(img * 64 + mm) * 64 + n0 + r] = f32_to_bf16(val);
  }
}

// ---------- 2x2 maxpool on conv2 output, flatten (c,h,w) -> x2 (4096,1024) bf16 ----------
__global__ void pool2_kernel(const unsigned short* __restrict__ y2, unsigned short* __restrict__ x2) {
  int idx = blockIdx.x * blockDim.x + threadIdx.x;
  if (idx >= 4096 * 1024) return;
  int pw = idx & 3, ph = (idx >> 2) & 3, c = (idx >> 4) & 63, img = idx >> 10;
  float mx = -1e30f;
#pragma unroll
  for (int dh = 0; dh < 2; dh++)
#pragma unroll
    for (int dw = 0; dw < 2; dw++) {
      int m = (2 * ph + dh) * 8 + (2 * pw + dw);
      mx = fmaxf(mx, bf16_to_f32(y2[(img * 64 + m) * 64 + c]));
    }
  x2[img * 1024 + c * 16 + ph * 4 + pw] = f32_to_bf16(mx);
}

// ---------- LDS-staged, double-buffered bf16 WMMA GEMM ----------
// out(MxN) = A(MxK) * W(NxK)^T + bias
// block tile 32(M) x 64(N), 8 waves (2x4 of 16x16 WMMA tiles), K-chunks of 32.
// Panels staged to LDS with GLOBAL_LOAD_ASYNC_TO_LDS (ASYNCcnt), double buffered
// so chunk i+1 copies overlap chunk i WMMA.
// grid = (N/64, M/32), block = 256.
__global__ __launch_bounds__(256) void gemm_bf16_lds_kernel(
    const unsigned short* __restrict__ A, const unsigned short* __restrict__ W,
    const float* __restrict__ bias, void* __restrict__ out,
    int K, int N, int relu, int out_bf16) {
  __shared__ unsigned short sA[2][32 * 32];   // 2 x 2KB
  __shared__ unsigned short sW[2][64 * 32];   // 2 x 4KB
  int tid = threadIdx.x;
  int m0 = blockIdx.y * 32, n0 = blockIdx.x * 64;

  // staging split: A panel = 256 threads x 8B (b64), W panel = 256 threads x 16B (b128)
  int arow = tid >> 3, acol = (tid & 7) * 4;   // 32 rows x 32 elems
  int wrow = tid >> 2, wcol = (tid & 3) * 8;   // 64 rows x 32 elems
  const unsigned short* agp = A + (size_t)(m0 + arow) * K + acol;
  const unsigned short* wgp = W + (size_t)(n0 + wrow) * K + wcol;

  auto stage = [&](int buf, int kb) {
#if USE_ASYNC_LDS
    __builtin_amdgcn_global_load_async_to_lds_b64(
        (g_i32x2*)(agp + kb),
        (l_i32x2*)&sA[buf][arow * 32 + acol], 0, 0);
    __builtin_amdgcn_global_load_async_to_lds_b128(
        (g_i32x4*)(wgp + kb),
        (l_i32x4*)&sW[buf][wrow * 32 + wcol], 0, 0);
#else
    *(unsigned long long*)&sA[buf][arow * 32 + acol] = *(const unsigned long long*)(agp + kb);
    *(u32x4*)&sW[buf][wrow * 32 + wcol] = *(const u32x4*)(wgp + kb);
#endif
  };

  int wv = tid >> 5, lane = tid & 31;
  int r = lane & 15, half = (lane < 16) ? 0 : 8;
  int mw = (wv >> 2) * 16;   // 0 / 16
  int nw = (wv & 3) * 16;    // 0..48

  int nch = K >> 5;
  stage(0, 0);
  v8f acc = {};
  for (int i = 0; i < nch; i++) {
    int buf = i & 1;
    if (i + 1 < nch) { stage(buf ^ 1, (i + 1) << 5); wait_async_le2(); }
    else             { wait_async_le0(); }
    __syncthreads();
    v16bf a = load_frag16((const __bf16*)&sA[buf][(mw + r) * 32 + half]);
    v16bf b = load_frag16((const __bf16*)&sW[buf][(nw + r) * 32 + half]);
    acc = __builtin_amdgcn_wmma_f32_16x16x32_bf16(false, a, false, b, (short)0, acc, false, false);
    __syncthreads();
  }

  int n = n0 + nw + r;
  float bv = bias ? bias[n] : 0.0f;
#pragma unroll
  for (int v = 0; v < 8; v++) {
    int m = m0 + mw + half + v;                // C layout: lane=N, vgpr=M (+8 hi lanes)
    float val = acc[v] + bv;
    if (relu) val = fmaxf(val, 0.0f);
    size_t o = (size_t)m * N + n;
    if (out_bf16) ((unsigned short*)out)[o] = f32_to_bf16(val);
    else          ((float*)out)[o] = val;
  }
}

// ---------- GRU-low: zero init ----------
__global__ void zero_h_kernel(float* __restrict__ h, unsigned short* __restrict__ hbf, int n) {
  int i = blockIdx.x * blockDim.x + threadIdx.x;
  if (i < n) { h[i] = 0.0f; hbf[i] = 0; }
}

// ---------- GRU-low: gate combine for step t ----------
__global__ void gru_low_combine_kernel(const float* __restrict__ gi_all, const float* __restrict__ gh,
                                       const int* __restrict__ partition, int t,
                                       float* __restrict__ h, unsigned short* __restrict__ hbf) {
  int idx = blockIdx.x * blockDim.x + threadIdx.x;   // 256*256
  int u = idx & 255, c = idx >> 8;
  int row = partition[c * 16 + t];
  const float* gi = gi_all + (size_t)row * 768;
  const float* gr = gh + (size_t)c * 768;
  float rg = sigmoidf_(gi[u] + gr[u]);
  float zg = sigmoidf_(gi[256 + u] + gr[256 + u]);
  float ng = tanhf(gi[512 + u] + rg * gr[512 + u]);
  float ho = h[idx];
  float hn = (1.0f - zg) * ng + zg * ho;
  h[idx] = hn;
  hbf[idx] = f32_to_bf16(hn);
}

// ---------- GRU-high: 256 sequential batch-1 steps, one persistent workgroup ----------
__global__ void gru_high_kernel(const float* __restrict__ cr,
                                const float* __restrict__ wi, const float* __restrict__ wh,
                                const float* __restrict__ bi, const float* __restrict__ bh,
                                float* __restrict__ sr) {
  __shared__ float h[256], xt[256], Gi[768], Gh[768];
  int tid = threadIdx.x;                 // 768 threads = 24 waves
  if (tid < 256) h[tid] = 0.0f;
  __syncthreads();
  const float* wri = wi + (size_t)tid * 256;
  const float* wrh = wh + (size_t)tid * 256;
  float bii = bi[tid], bhh = bh[tid];
  for (int t = 0; t < 256; t++) {
    if (tid < 256) xt[tid] = cr[t * 256 + tid];
    __syncthreads();
    float si = bii, sh = bhh;
    for (int k = 0; k < 256; k++) {
      si = fmaf(wri[k], xt[k], si);
      sh = fmaf(wrh[k], h[k], sh);
    }
    Gi[tid] = si; Gh[tid] = sh;
    __syncthreads();
    if (tid < 256) {
      float rg = sigmoidf_(Gi[tid] + Gh[tid]);
      float zg = sigmoidf_(Gi[256 + tid] + Gh[256 + tid]);
      float ng = tanhf(Gi[512 + tid] + rg * Gh[512 + tid]);
      h[tid] = (1.0f - zg) * ng + zg * h[tid];
    }
    __syncthreads();
  }
  if (tid < 256) sr[tid] = h[tid];
}

// ---------- cluster head: cr16 = relu(cr @ cluster_w^T + b) ----------
__global__ void cr16_kernel(const float* __restrict__ cr, const float* __restrict__ w,
                            const float* __restrict__ b, float* __restrict__ cr16) {
  int idx = blockIdx.x * blockDim.x + threadIdx.x;   // 256*16
  if (idx >= 4096) return;
  int j = idx & 15, c = idx >> 4;
  const float* x = cr + c * 256;
  const float* wr = w + j * 256;
  float s = b[j];
  for (int k = 0; k < 256; k++) s = fmaf(x[k], wr[k], s);
  cr16[idx] = fmaxf(s, 0.0f);
}

// ---------- state head + pair-bias precompute: base[o] = sr16 @ ws[o] + a1_b[o] ----------
__global__ void heads_kernel(const float* __restrict__ sr, const float* __restrict__ state_w,
                             const float* __restrict__ state_b, const float* __restrict__ a1_w,
                             const float* __restrict__ a1_b, float* __restrict__ base) {
  __shared__ float s16[16];
  int tid = threadIdx.x;                 // 32 threads
  if (tid < 16) {
    float s = state_b[tid];
    const float* wr = state_w + tid * 256;
    for (int k = 0; k < 256; k++) s = fmaf(sr[k], wr[k], s);
    s16[tid] = fmaxf(s, 0.0f);
  }
  __syncthreads();
  float bsum = a1_b[tid];
  const float* ws = a1_w + tid * 32;     // first 16 columns = ws
  for (int k = 0; k < 16; k++) bsum = fmaf(s16[k], ws[k], bsum);
  base[tid] = bsum;
}

// ---------- pairwise MLP: q[p] for all tril pairs ----------
__global__ void pairs_kernel(const float* __restrict__ cr16, const float* __restrict__ base,
                             const float* __restrict__ a1_w, const float* __restrict__ a2_w,
                             const float* __restrict__ a2_b, float* __restrict__ q, int np) {
  int p = blockIdx.x * blockDim.x + threadIdx.x;
  if (p >= np) return;
  int i = (int)((1.0 + sqrt(1.0 + 8.0 * (double)p)) * 0.5);
  while (i * (i - 1) / 2 > p) i--;
  while (i * (i + 1) / 2 <= p) i++;
  int j = p - i * (i - 1) / 2;
  float m16[16];
  const float* ci = cr16 + i * 16;
  const float* cj = cr16 + j * 16;
#pragma unroll
  for (int k = 0; k < 16; k++) m16[k] = ci[k] + cj[k];
  float acc = a2_b[0];
  for (int o = 0; o < 32; o++) {
    const float* wm = a1_w + o * 32 + 16;  // last 16 columns = wm
    float s = base[o];
#pragma unroll
    for (int k = 0; k < 16; k++) s = fmaf(m16[k], wm[k], s);
    acc = fmaf(fmaxf(s, 0.0f), a2_w[o], acc);
  }
  q[p] = acc;
}

// ---------- single-block softmax over 32640 ----------
__global__ void softmax_kernel(const float* __restrict__ q, float* __restrict__ out, int n) {
  __shared__ float red[1024];
  int tid = threadIdx.x;
  float mx = -1e30f;
  for (int i = tid; i < n; i += 1024) mx = fmaxf(mx, q[i]);
  red[tid] = mx;
  __syncthreads();
  for (int s = 512; s > 0; s >>= 1) { if (tid < s) red[tid] = fmaxf(red[tid], red[tid + s]); __syncthreads(); }
  float gmax = red[0];
  __syncthreads();
  float sum = 0.0f;
  for (int i = tid; i < n; i += 1024) sum += __expf(q[i] - gmax);
  red[tid] = sum;
  __syncthreads();
  for (int s = 512; s > 0; s >>= 1) { if (tid < s) red[tid] += red[tid + s]; __syncthreads(); }
  float inv = 1.0f / red[0];
  for (int i = tid; i < n; i += 1024) out[i] = __expf(q[i] - gmax) * inv;
}

// ==================== host ====================
extern "C" void kernel_launch(void* const* d_in, const int* in_sizes, int n_in,
                              void* d_out, int out_size, void* d_ws, size_t ws_size,
                              hipStream_t stream) {
  const float* images   = (const float*)d_in[0];
  const int*   part     = (const int*)  d_in[1];
  const float* conv1_w  = (const float*)d_in[2];
  const float* conv1_b  = (const float*)d_in[3];
  const float* conv2_w  = (const float*)d_in[4];
  const float* conv2_b  = (const float*)d_in[5];
  const float* fc_w     = (const float*)d_in[6];
  const float* fc_b     = (const float*)d_in[7];
  const float* gl_wi    = (const float*)d_in[8];
  const float* gl_wh    = (const float*)d_in[9];
  const float* gl_bi    = (const float*)d_in[10];
  const float* gl_bh    = (const float*)d_in[11];
  const float* gh_wi    = (const float*)d_in[12];
  const float* gh_wh    = (const float*)d_in[13];
  const float* gh_bi    = (const float*)d_in[14];
  const float* gh_bh    = (const float*)d_in[15];
  const float* state_w  = (const float*)d_in[16];
  const float* state_b  = (const float*)d_in[17];
  const float* clus_w   = (const float*)d_in[18];
  const float* clus_b   = (const float*)d_in[19];
  const float* a1_w     = (const float*)d_in[20];
  const float* a1_b     = (const float*)d_in[21];
  const float* a2_w     = (const float*)d_in[22];
  const float* a2_b     = (const float*)d_in[23];
  float* out = (float*)d_out;

  char* ws = (char*)d_ws;
  size_t off = 0;
  auto carve = [&](size_t bytes) -> void* {
    void* p = ws + off;
    off = (off + bytes + 255) & ~(size_t)255;
    return p;
  };
  unsigned short* x1_bf   = (unsigned short*)carve((size_t)4096 * 32 * 144 * 2);
  unsigned short* w2_bf   = (unsigned short*)carve((size_t)64 * 800 * 2);
  unsigned short* y2_bf   = (unsigned short*)carve((size_t)4096 * 64 * 64 * 2);
  unsigned short* x2_bf   = (unsigned short*)carve((size_t)4096 * 1024 * 2);
  unsigned short* wfc_bf  = (unsigned short*)carve((size_t)256 * 1024 * 2);
  unsigned short* feat_bf = (unsigned short*)carve((size_t)4096 * 256 * 2);
  unsigned short* wgi_bf  = (unsigned short*)carve((size_t)768 * 256 * 2);
  unsigned short* wgh_bf  = (unsigned short*)carve((size_t)768 * 256 * 2);
  float*          gi_all  = (float*)carve((size_t)4096 * 768 * 4);
  float*          h_f32   = (float*)carve((size_t)256 * 256 * 4);
  unsigned short* h_bf    = (unsigned short*)carve((size_t)256 * 256 * 2);
  float*          gh_buf  = (float*)carve((size_t)256 * 768 * 4);
  float*          sr      = (float*)carve(256 * 4);
  float*          cr16    = (float*)carve(4096 * 4);
  float*          base    = (float*)carve(32 * 4);
  float*          qbuf    = (float*)carve(32640 * 4);

  // weight conversions to bf16
  f2bf_kernel<<<(64 * 800 + 255) / 256, 256, 0, stream>>>(conv2_w, w2_bf, 64 * 800);
  f2bf_kernel<<<(256 * 1024 + 255) / 256, 256, 0, stream>>>(fc_w, wfc_bf, 256 * 1024);
  f2bf_kernel<<<(768 * 256 + 255) / 256, 256, 0, stream>>>(gl_wi, wgi_bf, 768 * 256);
  f2bf_kernel<<<(768 * 256 + 255) / 256, 256, 0, stream>>>(gl_wh, wgh_bf, 768 * 256);

  // conv1 + relu + pool
  conv1_pool_kernel<<<(4096 * 32 * 144 + 255) / 256, 256, 0, stream>>>(images, conv1_w, conv1_b, x1_bf);

  // conv2 implicit-GEMM WMMA + relu
  conv2_wmma_kernel<<<dim3(4, 4, 4096), 32, 0, stream>>>(x1_bf, w2_bf, conv2_b, y2_bf);

  // pool + flatten
  pool2_kernel<<<(4096 * 1024 + 255) / 256, 256, 0, stream>>>(y2_bf, x2_bf);

  // FC: feats = x2 @ fc_w^T + fc_b  (M=4096, K=1024, N=256) -> bf16
  gemm_bf16_lds_kernel<<<dim3(4, 128), 256, 0, stream>>>(x2_bf, wfc_bf, fc_b, feat_bf, 1024, 256, 0, 1);

  // GRU-low input gates: gi_all = feats @ gl_wi^T + gl_bi (M=4096, K=256, N=768) -> f32
  gemm_bf16_lds_kernel<<<dim3(12, 128), 256, 0, stream>>>(feat_bf, wgi_bf, gl_bi, gi_all, 256, 768, 0, 0);

  // GRU-low recurrence, 16 steps
  zero_h_kernel<<<(65536 + 255) / 256, 256, 0, stream>>>(h_f32, h_bf, 65536);
  for (int t = 0; t < 16; t++) {
    // gh = h @ gl_wh^T + gl_bh  (M=256, K=256, N=768) -> f32
    gemm_bf16_lds_kernel<<<dim3(12, 8), 256, 0, stream>>>(h_bf, wgh_bf, gl_bh, gh_buf, 256, 768, 0, 0);
    gru_low_combine_kernel<<<256, 256, 0, stream>>>(gi_all, gh_buf, part, t, h_f32, h_bf);
  }

  // GRU-high: 256 sequential batch-1 steps (latency-bound; persistent WG, LDS state)
  gru_high_kernel<<<1, 768, 0, stream>>>(h_f32, gh_wi, gh_wh, gh_bi, gh_bh, sr);

  // heads
  cr16_kernel<<<16, 256, 0, stream>>>(h_f32, clus_w, clus_b, cr16);
  heads_kernel<<<1, 32, 0, stream>>>(sr, state_w, state_b, a1_w, a1_b, base);

  // pairwise MLP over 32640 tril pairs
  pairs_kernel<<<(32640 + 255) / 256, 256, 0, stream>>>(cr16, base, a1_w, a2_w, a2_b, qbuf, 32640);

  // softmax
  softmax_kernel<<<1, 1024, 0, stream>>>(qbuf, out, 32640);
}
